// MABSA_Model_21981642621056
// MI455X (gfx1250) — matrix operand order repaired and checked by reading, not compiled, not run on hardware
//
#include <hip/hip_runtime.h>
#include <hip/hip_bf16.h>

#define N_NODES 4096
#define TDIM 768
#define IDIM 512
#define H 256
#define NHEADS 4
#define HDIM 64
#define NCLS 3

typedef __attribute__((ext_vector_type(16))) __bf16 v16bf;
typedef __attribute__((ext_vector_type(8)))  float v8f;
typedef __attribute__((ext_vector_type(4)))  unsigned int v4u;

union AFrag {
  v16bf v;
  unsigned short u[16];
  v4u q[2];
};

// round-to-nearest-even f32 -> bf16 via integer ops
__device__ __forceinline__ unsigned short f2bf(float f) {
  unsigned int u = __builtin_bit_cast(unsigned int, f);
  return (unsigned short)((u + 0x7FFFu + ((u >> 16) & 1u)) >> 16);
}

__device__ __forceinline__ v8f wmma_bf16(v16bf a, v16bf b, v8f c) {
  return __builtin_amdgcn_wmma_f32_16x16x32_bf16(false, a, false, b, (short)0, c,
                                                 false, false);
}

// A fragment (16x32 bf16, row-major source). ISA layout per lane: two
// contiguous 8-element runs (K {0..7} and {16..23}, +8 for upper half-wave).
// Both runs are 16B aligned for our strides -> two b128 loads.
__device__ __forceinline__ v16bf load_a_bf16(const unsigned short* A, int rowBase,
                                             int kBase, int lda) {
  const int lane = threadIdx.x & 31;
  const unsigned short* p =
      A + (size_t)(rowBase + (lane & 15)) * lda + kBase + (lane >> 4) * 8;
  AFrag a;
  a.q[0] = *(const v4u*)(p);       // K 0..7  (+half offset)
  a.q[1] = *(const v4u*)(p + 16);  // K 16..23 (+half offset)
  return a.v;
}

// B fragment from TRANSPOSED operand Bt [Nc, K] row-major:
// lane 0-15: col n=lane, K {0..7} and {16..23}; lane 16-31: +8 K offset.
// Two aligned b128 loads per lane.
__device__ __forceinline__ v16bf load_bT_bf16(const unsigned short* Bt, int kBase,
                                              int colBase, int ldk) {
  const int lane = threadIdx.x & 31;
  const unsigned short* p =
      Bt + (size_t)(colBase + (lane & 15)) * ldk + kBase + (lane >> 4) * 8;
  AFrag b;
  b.q[0] = *(const v4u*)(p);
  b.q[1] = *(const v4u*)(p + 16);
  return b.v;
}

// ---------------- elementwise / utility kernels ----------------

__global__ void cvt_f32_bf16(const float* __restrict__ in,
                             unsigned short* __restrict__ out, int n) {
  int i = blockIdx.x * blockDim.x + threadIdx.x;
  if (i < n) out[i] = f2bf(in[i]);
}

// out[n*K + k] = bf16(in[k*Nc + n])  (weight convert + transpose, done once)
__global__ void cvt_transpose_kernel(const float* __restrict__ in,
                                     unsigned short* __restrict__ out,
                                     int K, int Nc) {
  int i = blockIdx.x * blockDim.x + threadIdx.x;
  if (i >= K * Nc) return;
  int k = i / Nc, n = i % Nc;
  out[(size_t)n * K + k] = f2bf(in[i]);
}

__global__ void deg_init_kernel(float* deg, int n) {
  int i = blockIdx.x * blockDim.x + threadIdx.x;
  if (i < n) deg[i] = 1.0f;  // self loop
}

__global__ void deg_count_kernel(const int* __restrict__ dst, float* deg, int e) {
  int i = blockIdx.x * blockDim.x + threadIdx.x;
  if (i < e) atomicAdd(&deg[dst[i]], 1.0f);
}

__global__ void dinv_kernel(const float* __restrict__ deg, float* __restrict__ dinv,
                            int n) {
  int i = blockIdx.x * blockDim.x + threadIdx.x;
  if (i < n) dinv[i] = rsqrtf(deg[i]);
}

// out[n,c] = dinv[n]^2 * h[n,c]   (self-loop term; fully initializes out)
__global__ void gcn_self_kernel(const float* __restrict__ h,
                                const float* __restrict__ dinv,
                                float* __restrict__ out, int total) {
  int i = blockIdx.x * blockDim.x + threadIdx.x;
  if (i >= total) return;
  float di = dinv[i >> 8];
  out[i] = di * di * h[i];
}

__global__ void gcn_edge_kernel(const int* __restrict__ src,
                                const int* __restrict__ dst,
                                const float* __restrict__ dinv,
                                const float* __restrict__ h,
                                float* __restrict__ out, int e) {
  int i = blockIdx.x * blockDim.x + threadIdx.x;
  int eidx = i >> 6;  // 64 threads/edge, 4 channels each
  if (eidx >= e) return;
  int c4 = (i & 63) << 2;
  int s = src[eidx], d = dst[eidx];
  float nm = dinv[s] * dinv[d];
  const float4 hv = *(const float4*)(h + (size_t)s * H + c4);
  float* op = out + (size_t)d * H + c4;
  atomicAdd(op + 0, nm * hv.x);
  atomicAdd(op + 1, nm * hv.y);
  atomicAdd(op + 2, nm * hv.z);
  atomicAdd(op + 3, nm * hv.w);
}

__global__ void bias_act_kernel(const float* __restrict__ in,
                                const float* __restrict__ b,
                                float* __restrict__ outf,
                                unsigned short* __restrict__ outb,
                                int relu, int total) {
  int i = blockIdx.x * blockDim.x + threadIdx.x;
  if (i >= total) return;
  float v = in[i] + b[i & (H - 1)];
  if (relu) v = fmaxf(v, 0.0f);
  if (outf) outf[i] = v;
  if (outb) outb[i] = f2bf(v);
}

__global__ void classifier_kernel(const float* __restrict__ X,
                                  const float* __restrict__ Wc,
                                  const float* __restrict__ bc,
                                  float* __restrict__ out, int total) {
  int i = blockIdx.x * blockDim.x + threadIdx.x;
  if (i >= total) return;
  int n = i / NCLS, c = i % NCLS;
  float acc = bc[c];
  const float* xr = X + (size_t)n * H;
#pragma unroll 8
  for (int k = 0; k < H; ++k) acc += xr[k] * Wc[k * NCLS + c];
  out[i] = acc;
}

// ---------------- WMMA GEMM kernels ----------------

// C[M,Nc] = A[M,K] @ B[K,Nc] (+bias)(relu?), B given TRANSPOSED (Bt [Nc,K]).
// One 16x64 tile per wave (4 accumulators, A-fragment reused 4x).
__global__ __launch_bounds__(256) void wmma_gemm(
    const unsigned short* __restrict__ A, const unsigned short* __restrict__ Bt,
    const float* __restrict__ bias, float* __restrict__ Cf,
    unsigned short* __restrict__ Cb, unsigned short* __restrict__ CbT,
    int M, int Nc, int K, int relu) {
  const int wave = (blockIdx.x * blockDim.x + threadIdx.x) >> 5;
  const int ntn = Nc >> 6;  // 64-wide column tiles
  if (wave >= (M >> 4) * ntn) return;  // wave-uniform guard (EXEC stays full)
  const int tm = (wave / ntn) << 4;
  const int tn0 = (wave % ntn) << 6;
  v8f acc[4] = {{}, {}, {}, {}};
  for (int kk = 0; kk < K; kk += 32) {
    if (kk + 32 < K)
      __builtin_prefetch(A + (size_t)tm * K + kk + 32, 0, 1);  // global_prefetch_b8
    v16bf a = load_a_bf16(A, tm, kk, K);
#pragma unroll
    for (int t = 0; t < 4; ++t)
      acc[t] = wmma_bf16(a, load_bT_bf16(Bt, kk, tn0 + 16 * t, K), acc[t]);
  }
  const int lane = threadIdx.x & 31;
  const int rb = tm + ((lane >> 4) << 3);
#pragma unroll
  for (int t = 0; t < 4; ++t) {
    const int col = tn0 + 16 * t + (lane & 15);
    const float bv = bias ? bias[col] : 0.0f;
#pragma unroll
    for (int i = 0; i < 8; ++i) {
      float v = acc[t][i] + bv;
      if (relu) v = fmaxf(v, 0.0f);
      const int row = rb + i;
      if (Cf) Cf[(size_t)row * Nc + col] = v;
      if (Cb) Cb[(size_t)row * Nc + col] = f2bf(v);
      if (CbT) CbT[(size_t)col * M + row] = f2bf(v);  // transposed bf16 output
    }
  }
}

// combined = relu(Xt@Wt + bt) + relu(Xi@Wi + bi), bf16 out [N,H]; Wt/Wi transposed
__global__ __launch_bounds__(256) void proj_kernel(
    const unsigned short* __restrict__ Xt, const unsigned short* __restrict__ WtT,
    const float* __restrict__ bt, const unsigned short* __restrict__ Xi,
    const unsigned short* __restrict__ WiT, const float* __restrict__ bi,
    unsigned short* __restrict__ Cb) {
  const int wave = (blockIdx.x * blockDim.x + threadIdx.x) >> 5;
  const int ntn = H >> 6;  // 4
  if (wave >= (N_NODES >> 4) * ntn) return;
  const int tm = (wave / ntn) << 4;
  const int tn0 = (wave % ntn) << 6;
  v8f at[4] = {{}, {}, {}, {}};
  v8f ai[4] = {{}, {}, {}, {}};
  for (int kk = 0; kk < TDIM; kk += 32) {
    v16bf a = load_a_bf16(Xt, tm, kk, TDIM);
#pragma unroll
    for (int t = 0; t < 4; ++t)
      at[t] = wmma_bf16(a, load_bT_bf16(WtT, kk, tn0 + 16 * t, TDIM), at[t]);
  }
  for (int kk = 0; kk < IDIM; kk += 32) {
    v16bf a = load_a_bf16(Xi, tm, kk, IDIM);
#pragma unroll
    for (int t = 0; t < 4; ++t)
      ai[t] = wmma_bf16(a, load_bT_bf16(WiT, kk, tn0 + 16 * t, IDIM), ai[t]);
  }
  const int lane = threadIdx.x & 31;
  const int rb = tm + ((lane >> 4) << 3);
#pragma unroll
  for (int t = 0; t < 4; ++t) {
    const int col = tn0 + 16 * t + (lane & 15);
    const float btv = bt[col], biv = bi[col];
#pragma unroll
    for (int i = 0; i < 8; ++i) {
      float v = fmaxf(at[t][i] + btv, 0.0f) + fmaxf(ai[t][i] + biv, 0.0f);
      Cb[(size_t)(rb + i) * H + col] = f2bf(v);
    }
  }
}

// flash attention: one wave = one 16-query tile of one head; streams 32-key
// blocks. K[N,H] is already the transposed operand for Q.K^T; V is provided
// transposed (Vt [H,N]) so P.V B-fragments are contiguous too.
__global__ __launch_bounds__(256) void attn_kernel(
    const unsigned short* __restrict__ Q, const unsigned short* __restrict__ Km,
    const unsigned short* __restrict__ Vt, unsigned short* __restrict__ O) {
  __shared__ __align__(16) unsigned short plds[8][16 * 32];
  const int wv = threadIdx.x >> 5;
  const int wave = blockIdx.x * 8 + wv;
  const int head = wave >> 8;  // 256 q-tiles per head
  const int qrow = (wave & 255) << 4;
  const int hd0 = head * HDIM;
  const int lane = threadIdx.x & 31;
  const int lhalf = lane >> 4;
  const float scale = 0.125f;  // 1/sqrt(64)

  const v16bf qa0 = load_a_bf16(Q, qrow, hd0, H);
  const v16bf qa1 = load_a_bf16(Q, qrow, hd0 + 32, H);

  float m[8], l[8];
  v8f o0 = {}, o1 = {}, o2 = {}, o3 = {};
#pragma unroll
  for (int i = 0; i < 8; ++i) {
    m[i] = -1e30f;
    l[i] = 0.0f;
  }

  for (int kb = 0; kb < N_NODES; kb += 32) {
    // S[16x32] = Q Kb^T : K acts as Bt with ldk=H (contiguous head dim)
    v8f s0 = {}, s1 = {};
    s0 = wmma_bf16(qa0, load_bT_bf16(Km, hd0, kb, H), s0);
    s0 = wmma_bf16(qa1, load_bT_bf16(Km, hd0 + 32, kb, H), s0);
    s1 = wmma_bf16(qa0, load_bT_bf16(Km, hd0, kb + 16, H), s1);
    s1 = wmma_bf16(qa1, load_bT_bf16(Km, hd0 + 32, kb + 16, H), s1);

    float p0[8], p1[8], alpha[8];
#pragma unroll
    for (int i = 0; i < 8; ++i) {
      float a = s0[i] * scale, b = s1[i] * scale;
      float rm = fmaxf(a, b);
#pragma unroll
      for (int msk = 1; msk <= 8; msk <<= 1)
        rm = fmaxf(rm, __shfl_xor(rm, msk, 32));
      float mn = fmaxf(m[i], rm);
      alpha[i] = __expf(m[i] - mn);
      p0[i] = __expf(a - mn);
      p1[i] = __expf(b - mn);
      float rs = p0[i] + p1[i];
#pragma unroll
      for (int msk = 1; msk <= 8; msk <<= 1) rs += __shfl_xor(rs, msk, 32);
      l[i] = l[i] * alpha[i] + rs;
      m[i] = mn;
    }
    // stage P (C-layout) into LDS row-major 16x32, reload as A-fragment
#pragma unroll
    for (int i = 0; i < 8; ++i) {
      int row = i + (lhalf << 3);
      plds[wv][row * 32 + (lane & 15)] = f2bf(p0[i]);
      plds[wv][row * 32 + 16 + (lane & 15)] = f2bf(p1[i]);
    }
    __builtin_amdgcn_wave_barrier();
    AFrag pa;
    {
      const unsigned short* pp = &plds[wv][(lane & 15) * 32 + (lhalf << 3)];
      pa.q[0] = *(const v4u*)(pp);       // ds_load_b128
      pa.q[1] = *(const v4u*)(pp + 16);  // ds_load_b128
    }
    __builtin_amdgcn_wave_barrier();
#pragma unroll
    for (int i = 0; i < 8; ++i) {
      o0[i] *= alpha[i];
      o1[i] *= alpha[i];
      o2[i] *= alpha[i];
      o3[i] *= alpha[i];
    }
    o0 = wmma_bf16(pa.v, load_bT_bf16(Vt, kb, hd0 + 0, N_NODES), o0);
    o1 = wmma_bf16(pa.v, load_bT_bf16(Vt, kb, hd0 + 16, N_NODES), o1);
    o2 = wmma_bf16(pa.v, load_bT_bf16(Vt, kb, hd0 + 32, N_NODES), o2);
    o3 = wmma_bf16(pa.v, load_bT_bf16(Vt, kb, hd0 + 48, N_NODES), o3);
  }
#pragma unroll
  for (int i = 0; i < 8; ++i) {
    size_t row = (size_t)(qrow + i + (lhalf << 3)) * H + hd0 + (lane & 15);
    float inv = 1.0f / l[i];
    O[row + 0] = f2bf(o0[i] * inv);
    O[row + 16] = f2bf(o1[i] * inv);
    O[row + 32] = f2bf(o2[i] * inv);
    O[row + 48] = f2bf(o3[i] * inv);
  }
}

// ---------------- host orchestration ----------------

extern "C" void kernel_launch(void* const* d_in, const int* in_sizes, int n_in,
                              void* d_out, int out_size, void* d_ws, size_t ws_size,
                              hipStream_t stream) {
  const float* textf = (const float*)d_in[0];
  const float* imgf = (const float*)d_in[1];
  const int* edges = (const int*)d_in[2];
  const float* text_w = (const float*)d_in[3];
  const float* text_b = (const float*)d_in[4];
  const float* image_w = (const float*)d_in[5];
  const float* image_b = (const float*)d_in[6];
  const float* wq = (const float*)d_in[7];
  const float* bq = (const float*)d_in[8];
  const float* wk = (const float*)d_in[9];
  const float* bk = (const float*)d_in[10];
  const float* wv = (const float*)d_in[11];
  const float* bv = (const float*)d_in[12];
  const float* wo = (const float*)d_in[13];
  const float* bo = (const float*)d_in[14];
  const float* g1w = (const float*)d_in[15];
  const float* g1b = (const float*)d_in[16];
  const float* g2w = (const float*)d_in[17];
  const float* g2b = (const float*)d_in[18];
  const float* clw = (const float*)d_in[19];
  const float* clb = (const float*)d_in[20];

  const int E = in_sizes[2] / 2;
  const int* src = edges;
  const int* dst = edges + E;

  char* base = (char*)d_ws;
  size_t off = 0;
  auto alloc = [&](size_t bytes) -> void* {
    void* p = base + off;
    off += (bytes + 255) & ~(size_t)255;
    return p;
  };
  typedef unsigned short u16;
  u16* text_bf = (u16*)alloc((size_t)N_NODES * TDIM * 2);
  u16* img_bf = (u16*)alloc((size_t)N_NODES * IDIM * 2);
  u16* wtT = (u16*)alloc((size_t)TDIM * H * 2);   // [H, TDIM]
  u16* wiT = (u16*)alloc((size_t)IDIM * H * 2);   // [H, IDIM]
  u16* wqT = (u16*)alloc((size_t)H * H * 2);
  u16* wkT = (u16*)alloc((size_t)H * H * 2);
  u16* wvT = (u16*)alloc((size_t)H * H * 2);
  u16* woT = (u16*)alloc((size_t)H * H * 2);
  u16* g1wT = (u16*)alloc((size_t)H * H * 2);
  u16* g2wT = (u16*)alloc((size_t)H * H * 2);
  u16* comb_bf = (u16*)alloc((size_t)N_NODES * H * 2);
  u16* q_bf = (u16*)alloc((size_t)N_NODES * H * 2);
  u16* k_bf = (u16*)alloc((size_t)N_NODES * H * 2);
  u16* vT_bf = (u16*)alloc((size_t)N_NODES * H * 2);  // [H, N]
  u16* o_bf = (u16*)alloc((size_t)N_NODES * H * 2);
  u16* ao_bf = (u16*)alloc((size_t)N_NODES * H * 2);
  u16* g1_bf = (u16*)alloc((size_t)N_NODES * H * 2);
  float* hx = (float*)alloc((size_t)N_NODES * H * 4);
  float* agg = (float*)alloc((size_t)N_NODES * H * 4);
  float* hx2 = (float*)alloc((size_t)N_NODES * H * 4);
  float* agg2 = (float*)alloc((size_t)N_NODES * H * 4);
  float* g2f = (float*)alloc((size_t)N_NODES * H * 4);
  float* deg = (float*)alloc((size_t)N_NODES * 4);
  float* dinv = (float*)alloc((size_t)N_NODES * 4);

  auto cvt = [&](const float* in, u16* out, int n) {
    cvt_f32_bf16<<<(n + 255) / 256, 256, 0, stream>>>(in, out, n);
  };
  auto cvtT = [&](const float* in, u16* out, int K, int Nc) {
    cvt_transpose_kernel<<<(K * Nc + 255) / 256, 256, 0, stream>>>(in, out, K, Nc);
  };
  cvt(textf, text_bf, N_NODES * TDIM);
  cvt(imgf, img_bf, N_NODES * IDIM);
  cvtT(text_w, wtT, TDIM, H);
  cvtT(image_w, wiT, IDIM, H);
  cvtT(wq, wqT, H, H);
  cvtT(wk, wkT, H, H);
  cvtT(wv, wvT, H, H);
  cvtT(wo, woT, H, H);
  cvtT(g1w, g1wT, H, H);
  cvtT(g2w, g2wT, H, H);

  const int gemm_waves = (N_NODES / 16) * (H / 64);  // 1024
  const int gemm_blocks = gemm_waves / 8;            // 128

  // combined = relu(text@Wt+bt) + relu(img@Wi+bi)
  proj_kernel<<<gemm_blocks, 256, 0, stream>>>(text_bf, wtT, text_b, img_bf, wiT,
                                               image_b, comb_bf);
  // q/k row-major bf16, v transposed bf16 [H,N]
  wmma_gemm<<<gemm_blocks, 256, 0, stream>>>(comb_bf, wqT, bq, nullptr, q_bf,
                                             nullptr, N_NODES, H, H, 0);
  wmma_gemm<<<gemm_blocks, 256, 0, stream>>>(comb_bf, wkT, bk, nullptr, k_bf,
                                             nullptr, N_NODES, H, H, 0);
  wmma_gemm<<<gemm_blocks, 256, 0, stream>>>(comb_bf, wvT, bv, nullptr, nullptr,
                                             vT_bf, N_NODES, H, H, 0);
  // flash attention: 4 heads * 256 q-tiles = 1024 waves
  attn_kernel<<<128, 256, 0, stream>>>(q_bf, k_bf, vT_bf, o_bf);
  // output projection
  wmma_gemm<<<gemm_blocks, 256, 0, stream>>>(o_bf, woT, bo, nullptr, ao_bf,
                                             nullptr, N_NODES, H, H, 0);

  // degree / normalization (shared by both GCN layers)
  deg_init_kernel<<<(N_NODES + 255) / 256, 256, 0, stream>>>(deg, N_NODES);
  deg_count_kernel<<<(E + 255) / 256, 256, 0, stream>>>(dst, deg, E);
  dinv_kernel<<<(N_NODES + 255) / 256, 256, 0, stream>>>(deg, dinv, N_NODES);

  const int NH_total = N_NODES * H;
  const int edge_threads = E * 64;

  // GCN layer 1
  wmma_gemm<<<gemm_blocks, 256, 0, stream>>>(ao_bf, g1wT, nullptr, hx, nullptr,
                                             nullptr, N_NODES, H, H, 0);
  gcn_self_kernel<<<NH_total / 256, 256, 0, stream>>>(hx, dinv, agg, NH_total);
  gcn_edge_kernel<<<edge_threads / 256, 256, 0, stream>>>(src, dst, dinv, hx, agg, E);
  bias_act_kernel<<<NH_total / 256, 256, 0, stream>>>(agg, g1b, nullptr, g1_bf, 1,
                                                      NH_total);

  // GCN layer 2
  wmma_gemm<<<gemm_blocks, 256, 0, stream>>>(g1_bf, g2wT, nullptr, hx2, nullptr,
                                             nullptr, N_NODES, H, H, 0);
  gcn_self_kernel<<<NH_total / 256, 256, 0, stream>>>(hx2, dinv, agg2, NH_total);
  gcn_edge_kernel<<<edge_threads / 256, 256, 0, stream>>>(src, dst, dinv, hx2, agg2, E);
  bias_act_kernel<<<NH_total / 256, 256, 0, stream>>>(agg2, g2b, g2f, nullptr, 0,
                                                      NH_total);

  // classifier
  classifier_kernel<<<(N_NODES * NCLS + 255) / 256, 256, 0, stream>>>(
      g2f, clw, clb, (float*)d_out, N_NODES * NCLS);
}
